// GaussianRead_64201171141017
// MI455X (gfx1250) — matrix operand order repaired and checked by reading, not compiled, not run on hardware
//
#include <hip/hip_runtime.h>

typedef __attribute__((ext_vector_type(16))) _Float16 v16h;
typedef __attribute__((ext_vector_type(8)))  _Float16 v8h;
typedef __attribute__((ext_vector_type(8)))  float    v8f;

#define BSZ      512
#define TLEN     64
#define DH       256
#define MSLOTS   64
#define TILE_B   16
#define NTHREADS 256

// padded LDS strides (all byte strides multiples of 16 for b128 LDS access)
#define HS 260   // f32 h buffers
#define YS 260   // f32 wmma output
#define AS 272   // f16 A rows   (272*2 = 544 B)
#define WS 264   // f16 WuT rows (264*2 = 528 B)

// dynamic LDS layout (bytes)
//  f32: hA,hB (2*16*HS) | ynew (16*YS) | red (16*16*2) | We,be,bu,gamma,beta (5*256)
//  f16: WuT (256*WS) | Af (16*AS)
static constexpr size_t SMEM_F32 = (size_t)(2 * TILE_B * HS + TILE_B * YS + TILE_B * 16 * 2 + 5 * DH);
static constexpr size_t SMEM_F16 = (size_t)(DH * WS + TILE_B * AS);
static constexpr size_t SMEM_BYTES = SMEM_F32 * 4 + SMEM_F16 * 2;   // 200,960 B < 320 KB

__global__ __launch_bounds__(NTHREADS)
void memrnn_wmma_kernel(const float* __restrict__ x,   const float* __restrict__ We,
                        const float* __restrict__ be,  const float* __restrict__ Wu,
                        const float* __restrict__ bu,  const float* __restrict__ gamma,
                        const float* __restrict__ beta,const float* __restrict__ Wo,
                        const float* __restrict__ bo,  const float* __restrict__ ctx,
                        float* __restrict__ out)
{
    extern __shared__ char smem[];
    float*    hA   = (float*)smem;                  // h_{t-1} / h_{t-2} ping-pong
    float*    hB   = hA   + TILE_B * HS;
    float*    ynew = hB   + TILE_B * HS;            // WMMA output (pre-bias)
    float*    red  = ynew + TILE_B * YS;            // LN partial sums [row][grp][2]
    float*    WeL  = red  + TILE_B * 16 * 2;
    float*    beL  = WeL + DH;
    float*    buL  = beL + DH;
    float*    gaL  = buL + DH;
    float*    btL  = gaL + DH;
    _Float16* WuT  = (_Float16*)(btL + DH);         // Wu transposed: [n][k], k contiguous
    _Float16* Af   = WuT + DH * WS;                 // combined input rows, f16

    const int tid = threadIdx.x;
    const int b0  = blockIdx.x * TILE_B;

    // ---- one-time init: zero h state, stage params, transpose Wu -> f16 LDS ----
    for (int i = tid; i < 2 * TILE_B * HS; i += NTHREADS) hA[i] = 0.0f;
    if (tid < DH) {
        WeL[tid] = We[tid];  beL[tid] = be[tid];  buL[tid] = bu[tid];
        gaL[tid] = gamma[tid]; btL[tid] = beta[tid];
    }
    for (int i = tid; i < DH * DH; i += NTHREADS) {
        int k = i >> 8, n = i & 255;                // Wu[k][n] row-major
        WuT[n * WS + k] = (_Float16)Wu[i];
    }
    const float csg = 1.0f / (1.0f + expf(-ctx[0]));  // sigmoid(context_strength)
    __syncthreads();

    float* hp  = hA;   // h_{t-1}
    float* hpp = hB;   // h_{t-2}

    // elementwise mapping: 16 rows x 16 col-groups of 16
    const int r    = tid >> 4;
    const int c0   = (tid & 15) << 4;
    // wmma mapping: 8 waves, wave owns output cols [wave*32, wave*32+32)
    const int lane = tid & 31;
    const int half = lane >> 4;          // lane half (0: lanes 0-15, 1: lanes 16-31)
    const int lrow = lane & 15;
    const int n0   = (tid >> 5) << 5;    // wave * 32

    for (int t = 0; t < TLEN; ++t) {
        // ---- scalar attention weights (same for all batch rows; matches ref softmax) ----
        float e[5], S = 0.0f;
        #pragma unroll
        for (int j = 0; j < 5; ++j) {
            int   idx   = (t - 2 + j + MSLOTS) % MSLOTS;
            float delta = (float)idx - (float)t;
            e[j] = expf(-delta * delta * 0.125f);   // TEMP = 8
            S += e[j];
        }
        const float c1 = (t >= 1) ? e[1] / S : 0.0f;   // weight on h_{t-1}
        const float c2 = (t >= 2) ? e[0] / S : 0.0f;   // weight on h_{t-2}
        // (wrapped slots carry expf(-62^2/8)==0.0f in fp32, exactly as in the reference)

        // ---- phase 1: A rows = tanh(x*We+be) + cs*(c1*h1 + c2*h2) + h1  -> f16 ----
        const float xv = x[(b0 + r) * TLEN + t];
        #pragma unroll
        for (int i = 0; i < 16; ++i) {
            int   c  = c0 + i;
            float ip = tanhf(xv * WeL[c] + beL[c]);
            float h1 = hp[r * HS + c];
            float h2 = hpp[r * HS + c];
            Af[r * AS + c] = (_Float16)(ip + csg * (c1 * h1 + c2 * h2) + h1);
        }
        __syncthreads();

        // ---- phase 2: (16 x 256) @ (256 x 256) via v_wmma_f32_16x16x32_f16 ----
        v8f acc0 = {}; v8f acc1 = {};
        #pragma unroll
        for (int kb = 0; kb < 8; ++kb) {
            const int kbase = kb * 32;
            // A frag: lane half owns contiguous 16 K {h*16 .. h*16+15}
            const int ka = kbase + half * 16;
            v8h alo = *(const v8h*)&Af[lrow * AS + ka];
            v8h ahi = *(const v8h*)&Af[lrow * AS + ka + 8];
            v16h a  = __builtin_shufflevector(alo, ahi, 0,1,2,3,4,5,6,7,8,9,10,11,12,13,14,15);
            // B frag: lane holds column n, contiguous 16 K per lane half
            const int kbcol = kbase + half * 16;
            v16h bA = *(const v16h*)&WuT[(n0 + lrow) * WS + kbcol];
            v16h bB = *(const v16h*)&WuT[(n0 + 16 + lrow) * WS + kbcol];
            acc0 = __builtin_amdgcn_wmma_f32_16x16x32_f16(false, a, false, bA, (short)0, acc0, false, false);
            acc1 = __builtin_amdgcn_wmma_f32_16x16x32_f16(false, a, false, bB, (short)0, acc1, false, false);
        }
        // C/D layout: lane = n + 16*(m/8), vgpr = m%8
        {
            const int mbase = half * 8;
            const int ncol  = n0 + lrow;
            #pragma unroll
            for (int rr = 0; rr < 8; ++rr) {
                ynew[(mbase + rr) * YS + ncol]      = acc0[rr];
                ynew[(mbase + rr) * YS + ncol + 16] = acc1[rr];
            }
        }
        __syncthreads();

        // ---- phase 3: tanh + LayerNorm -> write h_t into hpp, then swap ----
        float g[16], sum = 0.0f, sq = 0.0f;
        #pragma unroll
        for (int i = 0; i < 16; ++i) {
            float y  = ynew[r * YS + c0 + i] + buL[c0 + i];
            float gg = tanhf(y);
            g[i] = gg; sum += gg; sq += gg * gg;
        }
        red[(r * 16 + (c0 >> 4)) * 2 + 0] = sum;
        red[(r * 16 + (c0 >> 4)) * 2 + 1] = sq;
        __syncthreads();
        float s = 0.0f, q = 0.0f;
        #pragma unroll
        for (int j = 0; j < 16; ++j) {
            s += red[(r * 16 + j) * 2 + 0];
            q += red[(r * 16 + j) * 2 + 1];
        }
        const float mu   = s * (1.0f / 256.0f);
        const float var  = q * (1.0f / 256.0f) - mu * mu;
        const float rinv = rsqrtf(var + 1e-5f);
        #pragma unroll
        for (int i = 0; i < 16; ++i) {
            int c = c0 + i;
            hpp[r * HS + c] = (g[i] - mu) * rinv * gaL[c] + btL[c];
        }
        __syncthreads();
        float* tmp = hp; hp = hpp; hpp = tmp;
    }

    // ---- final projection: out = h_T @ Wo + bo  (16 x 10 per block) ----
    if (tid < TILE_B * 10) {
        const int rr = tid / 10, cc = tid % 10;
        float acc = bo[cc];
        for (int k = 0; k < DH; ++k) acc += hp[rr * HS + k] * Wo[k * 10 + cc];
        out[(b0 + rr) * 10 + cc] = acc;
    }
}

extern "C" void kernel_launch(void* const* d_in, const int* in_sizes, int n_in,
                              void* d_out, int out_size, void* d_ws, size_t ws_size,
                              hipStream_t stream) {
    const float* x     = (const float*)d_in[0];
    const float* We    = (const float*)d_in[1];
    const float* be    = (const float*)d_in[2];
    const float* Wu    = (const float*)d_in[3];
    const float* bu    = (const float*)d_in[4];
    const float* gamma = (const float*)d_in[5];
    const float* beta  = (const float*)d_in[6];
    const float* Wo    = (const float*)d_in[7];
    const float* bo    = (const float*)d_in[8];
    const float* ctx   = (const float*)d_in[9];
    float* out = (float*)d_out;

    // allow > 64KB dynamic LDS (deterministic, idempotent; not a stream op)
    (void)hipFuncSetAttribute((const void*)memrnn_wmma_kernel,
                              hipFuncAttributeMaxDynamicSharedMemorySize,
                              (int)SMEM_BYTES);

    dim3 grid(BSZ / TILE_B), block(NTHREADS);
    memrnn_wmma_kernel<<<grid, block, SMEM_BYTES, stream>>>(
        x, We, be, Wu, bu, gamma, beta, Wo, bo, ctx, out);
}
